// VectorQuantizer_10222022165028
// MI455X (gfx1250) — matrix operand order repaired and checked by reading, not compiled, not run on hardware
//
#include <hip/hip_runtime.h>

typedef __attribute__((ext_vector_type(16))) _Float16 v16h;
typedef __attribute__((ext_vector_type(8)))  _Float16 v8h;
typedef __attribute__((ext_vector_type(8)))  float    v8f;

#define K_CODES 1024
#define DIM     256
#define N_VEC   32768        // 32 * 32 * 32 rows
#define X_ELEMS 8388608      // 32 * 256 * 32 * 32
#define ENC_ELEMS 33554432   // 32768 * 1024
#define ROWS    64           // rows per block

// ws layout:
//   [0,       524288): f16 codebook E16 [1024][256]
//   [524288,  528384): ck[1024] = ||e_k||^2 (f32)
//   [528384,  528388): global SSE accumulator (f32)

// ---------------------------------------------------------------------------
// Prep: f32 codebook -> f16 copy + per-code squared norms
// ---------------------------------------------------------------------------
__global__ void vq_prep(const float* __restrict__ emb,
                        _Float16* __restrict__ e16,
                        float* __restrict__ ck) {
  const int k = blockIdx.x;   // code
  const int c = threadIdx.x;  // channel
  float v = emb[k * DIM + c];
  e16[k * DIM + c] = (_Float16)v;
  float s = v * v;
  #pragma unroll
  for (int m = 16; m >= 1; m >>= 1) s += __shfl_down(s, m, 32);
  __shared__ float wsum[8];
  if ((threadIdx.x & 31) == 0) wsum[threadIdx.x >> 5] = s;
  __syncthreads();
  if (threadIdx.x == 0) {
    float t = 0.f;
    #pragma unroll
    for (int i = 0; i < 8; ++i) t += wsum[i];
    ck[k] = t;
  }
}

// ---------------------------------------------------------------------------
// Main fused kernel: 1 block = 64 rows of X. 512 threads = 16 waves.
// Wave w owns codes [w*64, w*64+64) and all 64 rows: 4 code-tiles x 4 row-
// tiles of v_wmma_f32_16x16x32_f16. Each B fragment (global, L2-resident)
// feeds 4 WMMAs -> 4x less L2 traffic than a 16-row block.
// ---------------------------------------------------------------------------
__global__ __launch_bounds__(512)
void vq_main(const float* __restrict__ x,
             const float* __restrict__ emb,
             const _Float16* __restrict__ e16,
             const float* __restrict__ ck,
             float* __restrict__ out,
             float* __restrict__ sse_acc) {
  __shared__ alignas(32) _Float16 ldsA[ROWS * DIM]; // 64 rows of X in f16 (32 KB)
  __shared__ float red_s[16 * ROWS];                // per-wave best score per row
  __shared__ int   red_c[16 * ROWS];                // per-wave best code per row
  __shared__ int   bestC[ROWS];
  __shared__ float blkSse;

  const int tid = threadIdx.x;
  const int n0  = blockIdx.x * ROWS;
  if (tid == 0) blkSse = 0.f;

  // ---- Phase 0: gather 64 rows (channel-major from [B,C,H,W]) into LDS f16
  #pragma unroll 4
  for (int i = 0; i < 32; ++i) {
    int idx = tid + i * 512;           // 0..16383 = 64 rows * 256 channels
    int row = idx & 63, c = idx >> 6;
    int n = n0 + row;
    int b = n >> 10, hw = n & 1023;
    ldsA[row * DIM + c] = (_Float16)x[b * (DIM * 1024) + c * 1024 + hw];
  }
  __syncthreads();

  // ---- Phase 1: S = X * E^T with WMMA (64 rows x 64 codes per wave)
  const int  wv = tid >> 5;
  const int  ln = tid & 31;
  const bool hi = ln >= 16;
  const int  lm = ln & 15;
  const int  codeBase = wv * 64;

  v8f acc[4][4] = {};                  // [code tile][row tile]
  #pragma unroll
  for (int kk = 0; kk < 8; ++kk) {
    const int k0 = kk * 32;
    // A 16x32 f16 fragments for 4 row tiles:
    // lanes<16 hold K 0..15, lanes>=16 K 16..31
    v16h a[4];
    #pragma unroll
    for (int rt = 0; rt < 4; ++rt) {
      const _Float16* ar = &ldsA[(rt * 16 + lm) * DIM + k0 + (hi ? 16 : 0)];
      union { v16h v; v8h h[2]; } au;
      au.h[0] = *(const v8h*)ar;
      au.h[1] = *(const v8h*)(ar + 8);
      a[rt] = au.v;
    }
    #pragma unroll
    for (int t = 0; t < 4; ++t) {
      // B 32x16 fragment: column = code, lanes<16 K 0..15, lanes>=16 K 16..31
      const _Float16* br = e16 + (codeBase + t * 16 + lm) * DIM + k0 + (hi ? 16 : 0);
      v16h bfrag = *(const v16h*)br;
      #pragma unroll
      for (int rt = 0; rt < 4; ++rt)
        acc[t][rt] = __builtin_amdgcn_wmma_f32_16x16x32_f16(
            false, a[rt], false, bfrag, (short)0, acc[t][rt], false, false);
    }
  }

  // ---- Phase 1b: per-wave argmin of (||e||^2 - 2*S) over 64 codes, per row
  float ckt[4];
  #pragma unroll
  for (int t = 0; t < 4; ++t) ckt[t] = ck[codeBase + t * 16 + lm];

  #pragma unroll
  for (int rt = 0; rt < 4; ++rt) {
    #pragma unroll
    for (int r = 0; r < 8; ++r) {      // C layout: lane holds rows r (+8 if hi)
      float bs = 3.4e38f; int bc = 0x7fffffff;
      #pragma unroll
      for (int t = 0; t < 4; ++t) {
        float s = ckt[t] - 2.0f * acc[t][rt][r];
        int   c = codeBase + t * 16 + lm;
        if (s < bs || (s == bs && c < bc)) { bs = s; bc = c; }
      }
      #pragma unroll
      for (int m = 1; m <= 8; m <<= 1) { // reduce within 16-lane half
        float os = __shfl_xor(bs, m, 32);
        int   oc = __shfl_xor(bc, m, 32);
        if (os < bs || (os == bs && oc < bc)) { bs = os; bc = oc; }
      }
      if (lm == 0) {
        int rowg = rt * 16 + r + (hi ? 8 : 0);
        red_s[wv * ROWS + rowg] = bs;
        red_c[wv * ROWS + rowg] = bc;
      }
    }
  }
  __syncthreads();

  // ---- Phase 2: cross-wave argmin; write one-hot encodings
  if (tid < ROWS) {
    float bs = red_s[tid]; int bc = red_c[tid];
    #pragma unroll
    for (int w = 1; w < 16; ++w) {
      float s = red_s[w * ROWS + tid]; int c = red_c[w * ROWS + tid];
      if (s < bs || (s == bs && c < bc)) { bs = s; bc = c; }
    }
    bestC[tid] = bc;
    out[(size_t)1 + X_ELEMS + (size_t)(n0 + tid) * K_CODES + bc] = 1.0f;
  }
  __syncthreads();

  // ---- Phase 3: quantized output in [B,C,H,W] order + SSE (x re-read, L2-hot)
  float lsum = 0.f;
  #pragma unroll 4
  for (int i = 0; i < 32; ++i) {
    int idx = tid + i * 512;
    int row = idx & 63, c = idx >> 6;
    int n = n0 + row;
    int b = n >> 10, hw = n & 1023;
    float q  = emb[bestC[row] * DIM + c];
    float xv = x[b * (DIM * 1024) + c * 1024 + hw];
    out[1 + b * (DIM * 1024) + c * 1024 + hw] = q;
    float d = q - xv;
    lsum += d * d;
  }
  #pragma unroll
  for (int m = 16; m >= 1; m >>= 1) lsum += __shfl_down(lsum, m, 32);
  if (ln == 0) atomicAdd(&blkSse, lsum);
  __syncthreads();
  if (tid == 0) atomicAdd(sse_acc, blkSse);
}

// loss = q_latent + 0.25 * e_latent = 1.25 * mean((q - x)^2)
__global__ void vq_finalize(const float* __restrict__ sse_acc,
                            float* __restrict__ out) {
  out[0] = 1.25f * sse_acc[0] / 8388608.0f;
}

extern "C" void kernel_launch(void* const* d_in, const int* in_sizes, int n_in,
                              void* d_out, int out_size, void* d_ws, size_t ws_size,
                              hipStream_t stream) {
  const float* x   = (const float*)d_in[0];   // [32,256,32,32] f32
  const float* emb = (const float*)d_in[1];   // [1024,256] f32
  float* out = (float*)d_out;                 // [loss | quantized | encodings]
  char*  ws  = (char*)d_ws;

  _Float16* e16 = (_Float16*)ws;              // 512 KB
  float*    ck  = (float*)(ws + 524288);      // 4 KB
  float*    acc = (float*)(ws + 528384);      // 4 B

  // zero the (mostly-zero) one-hot block and the SSE accumulator each call
  hipMemsetAsync(out + 1 + X_ELEMS, 0, (size_t)ENC_ELEMS * sizeof(float), stream);
  hipMemsetAsync(acc, 0, sizeof(float), stream);

  vq_prep<<<K_CODES, DIM, 0, stream>>>(emb, e16, ck);
  vq_main<<<N_VEC / ROWS, 512, 0, stream>>>(x, emb, e16, ck, out, acc);
  vq_finalize<<<1, 1, 0, stream>>>(acc, out);
}
